// MultiHeadAttention_34282428957190
// MI455X (gfx1250) — compile-verified, matrix operand order checked by reading, not run on hardware
//
#include <hip/hip_runtime.h>
#include <hip/hip_bf16.h>

#define B_  2
#define S_  2048
#define H_  1024
#define NH_ 16
#define HD_ 64
#define INV_SCALE 0.125f   // 1/sqrt(64)

typedef __attribute__((ext_vector_type(16))) __bf16 v16bf;
typedef __attribute__((ext_vector_type(8)))  __bf16 v8bf;
typedef __attribute__((ext_vector_type(8)))  float  v8f;
typedef __attribute__((ext_vector_type(4)))  int    v4i;

#define AS1_ __attribute__((address_space(1)))
#define AS3_ __attribute__((address_space(3)))

__device__ __forceinline__ v8f wmma_bf16(v16bf a, v16bf b, v8f c) {
    return __builtin_amdgcn_wmma_f32_16x16x32_bf16(
        false, a, false, b, (short)0, c, false, false);
}

// 16x32 fragment: per-lane two contiguous 8-elem (16B) groups at o0/o1.
__device__ __forceinline__ v16bf ld_frag(const __bf16* p, int o0, int o1) {
    v8bf lo = *(const v8bf*)(p + o0);
    v8bf hi = *(const v8bf*)(p + o1);
    v16bf r;
#pragma unroll
    for (int i = 0; i < 8; ++i) { r[i] = lo[i]; r[i + 8] = hi[i]; }
    return r;
}

// 16-byte global -> LDS async copy (CDNA5 path), with safe fallback.
__device__ __forceinline__ void async_cp16(void* lds, const void* g) {
#if defined(__HIP_DEVICE_COMPILE__) && \
    __has_builtin(__builtin_amdgcn_global_load_async_to_lds_b128)
    __builtin_amdgcn_global_load_async_to_lds_b128(
        (AS1_ v4i*)(g), (AS3_ v4i*)(lds), 0, 0);
#else
    *(v8bf*)lds = *(const v8bf*)g;
#endif
}

__device__ __forceinline__ void async_wait0() {
#if defined(__HIP_DEVICE_COMPILE__) && \
    __has_builtin(__builtin_amdgcn_s_wait_asynccnt)
    __builtin_amdgcn_s_wait_asynccnt(0);
#elif defined(__HIP_DEVICE_COMPILE__)
    asm volatile("s_wait_asynccnt 0" ::: "memory");
#endif
}

// ---------------------------------------------------------------------------
// Kernel 0: fp32 -> bf16 conversion (x, Wq, Wk, Wv, Wo), 8 elems/thread.
// ---------------------------------------------------------------------------
__global__ __launch_bounds__(256) void cvt_kernel(const float* __restrict__ src,
                                                  __bf16* __restrict__ dst)
{
    const size_t i = ((size_t)blockIdx.x * 256 + threadIdx.x) * 8;
    const float4 a = *(const float4*)(src + i);
    const float4 b = *(const float4*)(src + i + 4);
    v8bf o;
    o[0] = (__bf16)a.x; o[1] = (__bf16)a.y; o[2] = (__bf16)a.z; o[3] = (__bf16)a.w;
    o[4] = (__bf16)b.x; o[5] = (__bf16)b.y; o[6] = (__bf16)b.z; o[7] = (__bf16)b.w;
    *(v8bf*)(dst + i) = o;
}

// ---------------------------------------------------------------------------
// Kernel 1: QKV projection (x @ W.T) + RoPE. Block = 128(M) x 64(N = one head
// of one matrix). A and B tiles async-staged in LDS, shared by 8 waves.
// Wave = 16x64 (4 accumulators); RoPE partner (d,d+32) is c0<->c2, c1<->c3.
// V written transposed [B,NH,HD,S].
// ---------------------------------------------------------------------------
__global__ __launch_bounds__(256) void qkv_rope_kernel(
    const __bf16* __restrict__ xb, const float* __restrict__ cosb,
    const float* __restrict__ sinb,
    const __bf16* __restrict__ Wqb, const __bf16* __restrict__ Wkb,
    const __bf16* __restrict__ Wvb,
    __bf16* __restrict__ qbuf, __bf16* __restrict__ kbuf,
    __bf16* __restrict__ vtbuf)
{
    __shared__ __align__(16) __bf16 Ald[128][32];   // 8 KB
    __shared__ __align__(16) __bf16 Bld[64][32];    // 4 KB

    const int tid   = threadIdx.x;
    const int lane  = tid & 31, lo16 = lane & 15, khalf = lane >> 4;
    const int wi    = tid >> 5;

    const int nb  = blockIdx.x % 48;        // (mat, head)
    const int mb  = blockIdx.x / 48;
    const int mat = nb >> 4;                // 0=Q 1=K 2=V
    const int h   = nb & 15;
    const int m0  = mb * 128;
    const __bf16* Wsel = (mat == 0) ? Wqb : (mat == 1) ? Wkb : Wvb;
    const int n0 = h * HD_;

    v8f c[4] = {};
    for (int kb = 0; kb < H_; kb += 32) {
        __syncthreads();
        // A: 128 rows x 64B = 512 chunks (2/thread); B: 64 rows x 64B = 256.
        const int c0 = tid, c1 = tid + 256;
        async_cp16(&Ald[c0 >> 2][(c0 & 3) * 8],
                   xb + (size_t)(m0 + (c0 >> 2)) * H_ + kb + (c0 & 3) * 8);
        async_cp16(&Ald[c1 >> 2][(c1 & 3) * 8],
                   xb + (size_t)(m0 + (c1 >> 2)) * H_ + kb + (c1 & 3) * 8);
        async_cp16(&Bld[tid >> 2][(tid & 3) * 8],
                   Wsel + (size_t)(n0 + (tid >> 2)) * H_ + kb + (tid & 3) * 8);
        async_wait0();
        __syncthreads();

        v16bf a = ld_frag(&Ald[wi * 16 + lo16][0], khalf * 8, 16 + khalf * 8);
#pragma unroll
        for (int nt = 0; nt < 4; ++nt) {
            v16bf b = ld_frag(&Bld[nt * 16 + lo16][0],
                              khalf * 8, 16 + khalf * 8);
            c[nt] = wmma_bf16(a, b, c[nt]);
        }
    }

#pragma unroll
    for (int r = 0; r < 8; ++r) {
        const int mg = m0 + wi * 16 + r + 8 * khalf;
        const int b  = mg >> 11;
        const int s  = mg & (S_ - 1);
        const int bh = b * NH_ + h;
        if (mat < 2) {
            __bf16* dst = ((mat == 0) ? qbuf : kbuf) +
                          ((size_t)bh * S_ + s) * HD_;
#pragma unroll
            for (int p = 0; p < 2; ++p) {       // pairs (c0,c2), (c1,c3)
                const int d0 = p * 16 + lo16;
                const float v0 = c[p][r], v1 = c[p + 2][r];
                const float cs0 = cosb[s * HD_ + d0];
                const float sn0 = sinb[s * HD_ + d0];
                const float cs1 = cosb[s * HD_ + 32 + d0];
                const float sn1 = sinb[s * HD_ + 32 + d0];
                float r0 = v0 * cs0 - v1 * sn0;
                float r1 = v1 * cs1 + v0 * sn1;
                if (mat == 0) { r0 *= INV_SCALE; r1 *= INV_SCALE; }
                dst[d0]      = (__bf16)r0;
                dst[32 + d0] = (__bf16)r1;
            }
        } else {
            __bf16* dst = vtbuf + (size_t)bh * HD_ * S_;
#pragma unroll
            for (int nt = 0; nt < 4; ++nt)
                dst[(size_t)(nt * 16 + lo16) * S_ + s] = (__bf16)c[nt][r];
        }
    }
}

// ---------------------------------------------------------------------------
// Kernel 2: flash attention. Block = 8 waves sharing (b,h), covering 128 q
// rows. 64-key K/V tiles async-staged in LDS, shared by all waves. P goes
// C-layout -> A-layout through per-wave LDS staging.
// ---------------------------------------------------------------------------
__global__ __launch_bounds__(256) void attn_kernel(
    const __bf16* __restrict__ qbuf, const __bf16* __restrict__ kbuf,
    const __bf16* __restrict__ vtbuf, const float* __restrict__ mask,
    __bf16* __restrict__ attnbuf)
{
    __shared__ __align__(16) __bf16 Kld[64][64];     // 8 KB   [key][d]
    __shared__ __align__(16) __bf16 Vld[64][64];     // 8 KB   [d][key]
    __shared__ __align__(16) __bf16 Pld[8][16][64];  // 16 KB  per-wave P

    const int tid   = threadIdx.x;
    const int lane  = tid & 31, lo16 = lane & 15, khalf = lane >> 4;
    const int wi    = tid >> 5;

    const int qc = blockIdx.x & 15;
    const int bh = blockIdx.x >> 4;
    const int b  = bh >> 4, h = bh & 15;
    const int q0 = qc * 128 + wi * 16;

    const __bf16* Qp  = qbuf  + ((size_t)bh * S_ + q0) * HD_;
    const __bf16* Kp  = kbuf  + (size_t)bh * S_ * HD_;
    const __bf16* Vtp = vtbuf + (size_t)bh * HD_ * S_;
    const float*  M   = mask  + (size_t)b * S_ * S_;

    const __bf16* qrow = Qp + (size_t)lo16 * HD_;
    const v16bf qa0 = ld_frag(qrow, khalf * 8, 16 + khalf * 8);
    const v16bf qa1 = ld_frag(qrow, 32 + khalf * 8, 48 + khalf * 8);

    v8f co[4] = {};
    float mr[8], lr[8];
#pragma unroll
    for (int r = 0; r < 8; ++r) { mr[r] = -__builtin_inff(); lr[r] = 0.f; }

    for (int j0 = 0; j0 < S_; j0 += 64) {
        __syncthreads();
        // K tile: 64 keys x 128B = 512 chunks; V tile likewise. 4/thread.
        const int c0 = tid, c1 = tid + 256;
        async_cp16(&Kld[c0 >> 3][(c0 & 7) * 8],
                   Kp + (size_t)(j0 + (c0 >> 3)) * HD_ + (c0 & 7) * 8);
        async_cp16(&Kld[c1 >> 3][(c1 & 7) * 8],
                   Kp + (size_t)(j0 + (c1 >> 3)) * HD_ + (c1 & 7) * 8);
        async_cp16(&Vld[c0 >> 3][(c0 & 7) * 8],
                   Vtp + (size_t)(c0 >> 3) * S_ + j0 + (c0 & 7) * 8);
        async_cp16(&Vld[c1 >> 3][(c1 & 7) * 8],
                   Vtp + (size_t)(c1 >> 3) * S_ + j0 + (c1 & 7) * 8);
        async_wait0();
        __syncthreads();

        // --- scores: four 16x16 tiles over 64 keys (8 WMMAs) ---
        v8f st[4];
#pragma unroll
        for (int t = 0; t < 4; ++t) {
            const __bf16* krow = &Kld[t * 16 + lo16][0];
            v16bf bk0 = ld_frag(krow, khalf * 8, 16 + khalf * 8);
            v16bf bk1 = ld_frag(krow, 32 + khalf * 8, 48 + khalf * 8);
            v8f cc = {};
            cc = wmma_bf16(qa0, bk0, cc);
            cc = wmma_bf16(qa1, bk1, cc);
            st[t] = cc;
        }
        // --- mask add + block row max ---
        float tmax[8];
#pragma unroll
        for (int r = 0; r < 8; ++r) {
            const int qg = q0 + r + 8 * khalf;
            const float* mrow = M + (size_t)qg * S_ + j0;
            float mx = -__builtin_inff();
#pragma unroll
            for (int t = 0; t < 4; ++t) {
                st[t][r] += mrow[t * 16 + lo16];
                mx = fmaxf(mx, st[t][r]);
            }
            tmax[r] = mx;
        }
#pragma unroll
        for (int off = 1; off < 16; off <<= 1)
#pragma unroll
            for (int r = 0; r < 8; ++r)
                tmax[r] = fmaxf(tmax[r], __shfl_xor(tmax[r], off, 32));

        float alpha[8];
#pragma unroll
        for (int r = 0; r < 8; ++r) {
            const float mnew = fmaxf(mr[r], tmax[r]);
            alpha[r] = __expf(mr[r] - mnew);
            mr[r] = mnew;
        }
        // --- P = exp(S-m) -> per-wave LDS; row sums ---
        float lsum[8];
#pragma unroll
        for (int r = 0; r < 8; ++r) {
            float ls = 0.f;
            const int row = r + 8 * khalf;
#pragma unroll
            for (int t = 0; t < 4; ++t) {
                const float p = __expf(st[t][r] - mr[r]);
                Pld[wi][row][t * 16 + lo16] = (__bf16)p;
                ls += p;
            }
            lsum[r] = ls;
        }
#pragma unroll
        for (int off = 1; off < 16; off <<= 1)
#pragma unroll
            for (int r = 0; r < 8; ++r)
                lsum[r] += __shfl_xor(lsum[r], off, 32);
#pragma unroll
        for (int r = 0; r < 8; ++r) lr[r] = lr[r] * alpha[r] + lsum[r];

        asm volatile("s_wait_dscnt 0" ::: "memory");
        const __bf16* prow = &Pld[wi][lo16][0];
        const v16bf pa0 = ld_frag(prow, khalf * 8, 16 + khalf * 8);
        const v16bf pa1 = ld_frag(prow, 32 + khalf * 8, 48 + khalf * 8);

        // --- O = O*alpha + P @ V (8 WMMAs) ---
#pragma unroll
        for (int dt = 0; dt < 4; ++dt) {
#pragma unroll
            for (int r = 0; r < 8; ++r) co[dt][r] *= alpha[r];
            const __bf16* vrow = &Vld[dt * 16 + lo16][0];
            v16bf bv0 = ld_frag(vrow, khalf * 8, 16 + khalf * 8);
            v16bf bv1 = ld_frag(vrow, 32 + khalf * 8, 48 + khalf * 8);
            co[dt] = wmma_bf16(pa0, bv0, co[dt]);
            co[dt] = wmma_bf16(pa1, bv1, co[dt]);
        }
    }

#pragma unroll
    for (int r = 0; r < 8; ++r) {
        const float inv = 1.0f / lr[r];
        const int sg = q0 + r + 8 * khalf;
        __bf16* dst = attnbuf + ((size_t)b * S_ + sg) * H_ + h * HD_;
#pragma unroll
        for (int dt = 0; dt < 4; ++dt)
            dst[dt * 16 + lo16] = (__bf16)(co[dt][r] * inv);
    }
}

// ---------------------------------------------------------------------------
// Kernel 3: output projection attn @ Wo.T -> fp32. Same 128x64 LDS-tiled GEMM.
// ---------------------------------------------------------------------------
__global__ __launch_bounds__(256) void out_proj_kernel(
    const __bf16* __restrict__ attn, const __bf16* __restrict__ Wob,
    float* __restrict__ out)
{
    __shared__ __align__(16) __bf16 Ald[128][32];
    __shared__ __align__(16) __bf16 Bld[64][32];

    const int tid   = threadIdx.x;
    const int lane  = tid & 31, lo16 = lane & 15, khalf = lane >> 4;
    const int wi    = tid >> 5;

    const int nb = blockIdx.x & 15;
    const int mb = blockIdx.x >> 4;
    const int m0 = mb * 128, n0 = nb * 64;

    v8f c[4] = {};
    for (int kb = 0; kb < H_; kb += 32) {
        __syncthreads();
        const int c0 = tid, c1 = tid + 256;
        async_cp16(&Ald[c0 >> 2][(c0 & 3) * 8],
                   attn + (size_t)(m0 + (c0 >> 2)) * H_ + kb + (c0 & 3) * 8);
        async_cp16(&Ald[c1 >> 2][(c1 & 3) * 8],
                   attn + (size_t)(m0 + (c1 >> 2)) * H_ + kb + (c1 & 3) * 8);
        async_cp16(&Bld[tid >> 2][(tid & 3) * 8],
                   Wob + (size_t)(n0 + (tid >> 2)) * H_ + kb + (tid & 3) * 8);
        async_wait0();
        __syncthreads();

        v16bf a = ld_frag(&Ald[wi * 16 + lo16][0], khalf * 8, 16 + khalf * 8);
#pragma unroll
        for (int nt = 0; nt < 4; ++nt) {
            v16bf b = ld_frag(&Bld[nt * 16 + lo16][0],
                              khalf * 8, 16 + khalf * 8);
            c[nt] = wmma_bf16(a, b, c[nt]);
        }
    }
#pragma unroll
    for (int r = 0; r < 8; ++r) {
        const int mg = m0 + wi * 16 + r + 8 * khalf;
#pragma unroll
        for (int nt = 0; nt < 4; ++nt)
            out[(size_t)mg * H_ + n0 + nt * 16 + lo16] = c[nt][r];
    }
}

// ---------------------------------------------------------------------------
extern "C" void kernel_launch(void* const* d_in, const int* in_sizes, int n_in,
                              void* d_out, int out_size, void* d_ws,
                              size_t ws_size, hipStream_t stream)
{
    const float* x    = (const float*)d_in[0];
    const float* mask = (const float*)d_in[1];
    const float* cosb = (const float*)d_in[2];
    const float* sinb = (const float*)d_in[3];
    const float* Wq   = (const float*)d_in[4];
    const float* Wk   = (const float*)d_in[5];
    const float* Wv   = (const float*)d_in[6];
    const float* Wo   = (const float*)d_in[7];
    float* out = (float*)d_out;

    const size_t M4 = (size_t)4 * 1024 * 1024;   // 4M elems
    const size_t M1 = (size_t)1024 * 1024;       // 1M elems
    __bf16* xb   = (__bf16*)d_ws;                // [4096,1024]
    __bf16* Wqb  = xb + M4;
    __bf16* Wkb  = Wqb + M1;
    __bf16* Wvb  = Wkb + M1;
    __bf16* Wob  = Wvb + M1;
    __bf16* qb   = Wob + M1;                     // [B,NH,S,HD]
    __bf16* kb   = qb + M4;
    __bf16* vtb  = kb + M4;                      // [B,NH,HD,S]
    __bf16* attn = vtb + M4;                     // [B,S,H]

    // fp32 -> bf16 pre-pass
    cvt_kernel<<<2048, 256, 0, stream>>>(x, xb);
    cvt_kernel<<<512, 256, 0, stream>>>(Wq, Wqb);
    cvt_kernel<<<512, 256, 0, stream>>>(Wk, Wkb);
    cvt_kernel<<<512, 256, 0, stream>>>(Wv, Wvb);
    cvt_kernel<<<512, 256, 0, stream>>>(Wo, Wob);

    // K1: 32 m-blocks x (3 mats x 16 heads) = 1536 blocks
    qkv_rope_kernel<<<1536, 256, 0, stream>>>(xb, cosb, sinb, Wqb, Wkb, Wvb,
                                              qb, kb, vtb);
    // K2: (B*NH) x (S/128) = 512 blocks
    attn_kernel<<<512, 256, 0, stream>>>(qb, kb, vtb, mask, attn);
    // K3: 32 m-blocks x 16 n-blocks = 512 blocks
    out_proj_kernel<<<512, 256, 0, stream>>>(attn, Wob, out);
}
